// MoELayer_82205674045696
// MI455X (gfx1250) — compile-verified, matrix operand order checked by reading, not run on hardware
//
#include <hip/hip_runtime.h>
#include <hip/hip_bf16.h>

// ---------------------------------------------------------------------------
// MoE (B=2, S=2048, D=512, H=1024, E=8, K=2) for gfx1250 (MI455X, wave32)
// bf16 WMMA pipeline: router -> cast/transpose -> per-expert GEMM/SwiGLU/GEMM
// GEMM: 128x128 block tile, BK=32, 4 waves @ 64x64, double-buffered LDS,
//       async global->LDS loads (GLOBAL_LOAD_ASYNC_TO_LDS_B128 / ASYNCcnt).
// ---------------------------------------------------------------------------

#define MOE_S   4096   // B*S tokens
#define MOE_D   512
#define MOE_H   1024
#define MOE_E   8

typedef __attribute__((ext_vector_type(16))) __bf16 v16bf;
typedef __attribute__((ext_vector_type(8)))  __bf16 v8bf;
typedef __attribute__((ext_vector_type(4)))  __bf16 v4bf;
typedef __attribute__((ext_vector_type(8)))  float  v8f;
typedef __attribute__((ext_vector_type(4)))  float  v4f;

union Frag {
    v16bf v;
    v8bf  h[2];
};

// ---- CDNA5 async global->LDS path (guarded; falls back to reg staging) ----
#if defined(__gfx1250__) && \
    __has_builtin(__builtin_amdgcn_global_load_async_to_lds_b128) && \
    __has_builtin(__builtin_amdgcn_s_wait_asynccnt)
#define USE_ASYNC_LDS 1
#else
#define USE_ASYNC_LDS 0
#endif

#if USE_ASYNC_LDS
// Builtin signature (discovered via clang diagnostics):
//   void __builtin_amdgcn_global_load_async_to_lds_b128(
//       int4* gsrc, __shared__ int4* ldst, imm int offset, imm int cpol)
typedef int gv4i __attribute__((__vector_size__(4 * sizeof(int))));
typedef __attribute__((address_space(3))) gv4i as3_gv4i;

__device__ __forceinline__ as3_gv4i* to_lds(const void* p) {
    // Generic->LDS: low 32 bits of a generic LDS address are the DS address
    // (ISA 10.2 aperture mapping).
    return (as3_gv4i*)(unsigned int)(unsigned long long)p;
}
__device__ __forceinline__ gv4i* to_gv4i(const void* p) {
    // integer round-trip strips const; generic pointer, int4 pointee
    return (gv4i*)(unsigned long long)p;
}
#endif

// ---------------------------------------------------------------------------
// Router: one wave (32 lanes) per token. 8 dot products over D=512,
// wave-reduce, top-2 (first-index tie break, like lax.top_k), softmax over
// the two logits, scatter into dense combine matrix cw[token][8].
// ---------------------------------------------------------------------------
__global__ void router_kernel(const float* __restrict__ x,
                              const float* __restrict__ rw,
                              float* __restrict__ cw) {
    const int token = blockIdx.x;
    const int lane  = threadIdx.x;          // 0..31
    const float* xr = x + (size_t)token * MOE_D;

    float acc[MOE_E];
#pragma unroll
    for (int e = 0; e < MOE_E; ++e) acc[e] = 0.0f;

    for (int d = lane; d < MOE_D; d += 32) {
        const float xv = xr[d];
#pragma unroll
        for (int e = 0; e < MOE_E; ++e)
            acc[e] += xv * rw[(size_t)e * MOE_D + d];
    }
#pragma unroll
    for (int e = 0; e < MOE_E; ++e) {
        for (int off = 16; off > 0; off >>= 1)
            acc[e] += __shfl_xor(acc[e], off, 32);
    }
    if (lane == 0) {
        int   i1 = 0;  float v1 = acc[0];
        for (int e = 1; e < MOE_E; ++e)
            if (acc[e] > v1) { v1 = acc[e]; i1 = e; }
        int   i2 = -1; float v2 = -__builtin_huge_valf();
        for (int e = 0; e < MOE_E; ++e)
            if (e != i1 && acc[e] > v2) { v2 = acc[e]; i2 = e; }
        // softmax over {v1, v2}, v1 is the max
        const float e2  = __expf(v2 - v1);
        const float inv = 1.0f / (1.0f + e2);
        const float w1  = inv;
        const float w2  = e2 * inv;
#pragma unroll
        for (int e = 0; e < MOE_E; ++e) {
            float w = (e == i1) ? w1 : ((e == i2) ? w2 : 0.0f);
            cw[(size_t)token * MOE_E + e] = w;
        }
    }
}

// ---------------------------------------------------------------------------
// fp32 -> bf16 cast, 4 elements per thread
// ---------------------------------------------------------------------------
__global__ void cast_bf16_kernel(const float* __restrict__ src,
                                 __bf16* __restrict__ dst, size_t n4) {
    size_t i = (size_t)blockIdx.x * blockDim.x + threadIdx.x;
    if (i >= n4) return;
    v4f f = ((const v4f*)src)[i];
    ((v4bf*)dst)[i] = __builtin_convertvector(f, v4bf);
}

// ---------------------------------------------------------------------------
// Batched LDS-tiled transpose + cast: src[R][C] fp32 -> dst[C][R] bf16.
// grid.z = batch (expert). Coalesced on both sides. R, C multiples of 32.
// ---------------------------------------------------------------------------
__global__ void transpose_cast_kernel(const float* __restrict__ src,
                                      __bf16* __restrict__ dst,
                                      int R, int C,
                                      size_t sbatch, size_t dbatch) {
    __shared__ float tile[32][33];
    const float* s = src + (size_t)blockIdx.z * sbatch;
    __bf16*      d = dst + (size_t)blockIdx.z * dbatch;
    const int c0 = blockIdx.x * 32;
    const int r0 = blockIdx.y * 32;
    const int tx = threadIdx.x;   // 0..31
    const int ty = threadIdx.y;   // 0..7
#pragma unroll
    for (int i = 0; i < 32; i += 8)
        tile[ty + i][tx] = s[(size_t)(r0 + ty + i) * C + (c0 + tx)];
    __syncthreads();
#pragma unroll
    for (int i = 0; i < 32; i += 8)
        d[(size_t)(c0 + ty + i) * R + (r0 + tx)] = (__bf16)tile[tx][ty + i];
}

// ---------------------------------------------------------------------------
// SwiGLU: Z[s][0:H] = gate, Z[s][H:2H] = up ; act = silu(g) * u  (bf16)
// ---------------------------------------------------------------------------
__global__ void swiglu_kernel(const __bf16* __restrict__ Z,
                              __bf16* __restrict__ act) {
    size_t i = (size_t)blockIdx.x * blockDim.x + threadIdx.x;
    const size_t n = (size_t)MOE_S * MOE_H;
    if (i >= n) return;
    const size_t srow = i / MOE_H;
    const size_t h    = i - srow * MOE_H;
    const float g = (float)Z[srow * (2 * MOE_H) + h];
    const float u = (float)Z[srow * (2 * MOE_H) + MOE_H + h];
    const float sg = g / (1.0f + __expf(-g));
    act[i] = (__bf16)(sg * u);
}

// ---------------------------------------------------------------------------
// bf16 WMMA GEMM:  C[M][N] = A[M][K] * Bt[N][K]^T   (both row-major, K-contig)
// 128x128 block tile, BK=32. 128 threads = 4 waves, each wave 64x64:
// 16 v_wmma_f32_16x16x32_bf16 per wave per K-step vs 16 ds_load_b128 (1:1).
// Double-buffered LDS; global traffic for step k+1 overlaps compute of k.
// MODE 0: store bf16 into Zout[M][N]
// MODE 1: Out[M][N] (+)= cw[row*8+expert] * acc   (fp32 epilogue)
// ---------------------------------------------------------------------------
#define BM  128
#define BN  128
#define BK  32
#define LDT 40   // padded LDS row stride (elements); 80B rows, 16B aligned

template <int MODE>
__global__ __launch_bounds__(128)
void gemm_bf16_kernel(const __bf16* __restrict__ A,
                      const __bf16* __restrict__ Bt,
                      int M, int N, int Kdim,
                      __bf16* __restrict__ Zout,
                      float* __restrict__ Out,
                      const float* __restrict__ cw,
                      int expert, int accumulate) {
    __shared__ __bf16 As[2][BM * LDT];
    __shared__ __bf16 Bs[2][BN * LDT];

    const int tid  = threadIdx.x;
    const int wave = tid >> 5;           // 0..3
    const int lane = tid & 31;
    const int wm   = (wave >> 1) * 64;   // wave M offset (2 waves in M)
    const int wn   = (wave & 1)  * 64;   // wave N offset (2 waves in N)
    const int hs   = lane >> 4;          // K-half select per fragment layout
    const int lr   = lane & 15;          // row-within-16 / column lane

    const int m0 = blockIdx.y * BM;
    const int n0 = blockIdx.x * BN;

    // Per-thread 16B chunk coords: 512 chunks per tile, 4 per thread.
    int crow[4], ck8[4];
#pragma unroll
    for (int i = 0; i < 4; ++i) {
        const int c = tid + i * 128;
        crow[i] = c >> 2;          // 0..127
        ck8[i]  = (c & 3) * 8;     // 0,8,16,24 (elements)
    }

    v8f acc[4][4];
#pragma unroll
    for (int i = 0; i < 4; ++i)
#pragma unroll
        for (int j = 0; j < 4; ++j)
            acc[i][j] = (v8f){0.f, 0.f, 0.f, 0.f, 0.f, 0.f, 0.f, 0.f};

    const int nsteps = Kdim / BK;

    auto compute = [&](int p) {
        Frag a[4], b[4];
#pragma unroll
        for (int i = 0; i < 4; ++i) {
            const int r = wm + i * 16 + lr;
            a[i].h[0] = *(const v8bf*)&As[p][r * LDT + hs * 8];
            a[i].h[1] = *(const v8bf*)&As[p][r * LDT + 16 + hs * 8];
        }
#pragma unroll
        for (int j = 0; j < 4; ++j) {
            const int r = wn + j * 16 + lr;
            b[j].h[0] = *(const v8bf*)&Bs[p][r * LDT + hs * 8];
            b[j].h[1] = *(const v8bf*)&Bs[p][r * LDT + 16 + hs * 8];
        }
#pragma unroll
        for (int i = 0; i < 4; ++i)
#pragma unroll
            for (int j = 0; j < 4; ++j)
                acc[i][j] = __builtin_amdgcn_wmma_f32_16x16x32_bf16(
                    false, a[i].v, false, b[j].v,
                    (short)0, acc[i][j], false, false);
    };

#if USE_ASYNC_LDS
    // ---- async global->LDS double buffer (ASYNCcnt-tracked, no reg staging)
    auto stage_async = [&](int p, int kt) {
#pragma unroll
        for (int i = 0; i < 4; ++i) {
            __builtin_amdgcn_global_load_async_to_lds_b128(
                to_gv4i(&A[(size_t)(m0 + crow[i]) * Kdim + kt + ck8[i]]),
                to_lds(&As[p][crow[i] * LDT + ck8[i]]), 0, 0);
            __builtin_amdgcn_global_load_async_to_lds_b128(
                to_gv4i(&Bt[(size_t)(n0 + crow[i]) * Kdim + kt + ck8[i]]),
                to_lds(&Bs[p][crow[i] * LDT + ck8[i]]), 0, 0);
        }
    };
    stage_async(0, 0);
    __builtin_amdgcn_s_wait_asynccnt(0);
    __syncthreads();
    for (int s = 0; s < nsteps; ++s) {
        const int p = s & 1;
        if (s + 1 < nsteps) stage_async(p ^ 1, (s + 1) * BK);   // overlap
        compute(p);
        __builtin_amdgcn_s_wait_asynccnt(0);
        __syncthreads();
    }
#else
    // ---- register-staged double buffer
    v8bf ar[4], br[4];
    auto gload = [&](int kt) {
#pragma unroll
        for (int i = 0; i < 4; ++i) {
            ar[i] = *(const v8bf*)&A [(size_t)(m0 + crow[i]) * Kdim + kt + ck8[i]];
            br[i] = *(const v8bf*)&Bt[(size_t)(n0 + crow[i]) * Kdim + kt + ck8[i]];
        }
    };
    auto lstore = [&](int p) {
#pragma unroll
        for (int i = 0; i < 4; ++i) {
            *(v8bf*)&As[p][crow[i] * LDT + ck8[i]] = ar[i];
            *(v8bf*)&Bs[p][crow[i] * LDT + ck8[i]] = br[i];
        }
    };
    gload(0);
    lstore(0);
    __syncthreads();
    for (int s = 0; s < nsteps; ++s) {
        const int p = s & 1;
        if (s + 1 < nsteps) gload((s + 1) * BK);   // in flight during compute
        compute(p);
        if (s + 1 < nsteps) { lstore(p ^ 1); __syncthreads(); }
    }
#endif

    // ---- epilogue; C/D layout: VGPR r, lane -> (row = r + 8*hs, col = lr)
#pragma unroll
    for (int i = 0; i < 4; ++i) {
#pragma unroll
        for (int j = 0; j < 4; ++j) {
#pragma unroll
            for (int r = 0; r < 8; ++r) {
                const int grow = m0 + wm + i * 16 + r + hs * 8;
                const int gcol = n0 + wn + j * 16 + lr;
                const float v  = acc[i][j][r];
                if (MODE == 0) {
                    Zout[(size_t)grow * N + gcol] = (__bf16)v;
                } else {
                    const float  scale = cw[(size_t)grow * MOE_E + expert];
                    const size_t o     = (size_t)grow * N + gcol;
                    const float  prev  = accumulate ? Out[o] : 0.0f;
                    Out[o] = prev + scale * v;
                }
            }
        }
    }
}

// ---------------------------------------------------------------------------
// Launcher
// ---------------------------------------------------------------------------
extern "C" void kernel_launch(void* const* d_in, const int* in_sizes, int n_in,
                              void* d_out, int out_size, void* d_ws, size_t ws_size,
                              hipStream_t stream) {
    (void)in_sizes; (void)n_in; (void)out_size; (void)ws_size;

    const float* x  = (const float*)d_in[0];   // [2,2048,512]
    const float* rw = (const float*)d_in[1];   // [8,512]
    const float* wg = (const float*)d_in[2];   // [8,512,1024]
    const float* wu = (const float*)d_in[3];   // [8,512,1024]
    const float* wd = (const float*)d_in[4];   // [8,1024,512]
    float* out      = (float*)d_out;           // [2,2048,512]

    // workspace carve-up (16B aligned blocks)
    char* ws = (char*)d_ws;
    float*  cw    = (float*)ws;  ws += (size_t)MOE_S * MOE_E * sizeof(float);        // 128 KB
    __bf16* xb    = (__bf16*)ws; ws += (size_t)MOE_S * MOE_D * 2;                    // 4 MB
    __bf16* wtGU  = (__bf16*)ws; ws += (size_t)MOE_E * 2 * MOE_H * MOE_D * 2;        // 16 MB  [E][2H][D]
    __bf16* wtDN  = (__bf16*)ws; ws += (size_t)MOE_E * MOE_D * MOE_H * 2;            // 8 MB   [E][D][H]
    __bf16* Z     = (__bf16*)ws; ws += (size_t)MOE_S * 2 * MOE_H * 2;                // 16 MB  [S][2H]
    __bf16* act   = (__bf16*)ws; ws += (size_t)MOE_S * MOE_H * 2;                    // 8 MB   [S][H]

    // 1) router -> combine weights
    router_kernel<<<MOE_S, 32, 0, stream>>>(x, rw, cw);

    // 2) cast activations; transpose+cast weights into K-contiguous bf16
    {
        const size_t n4 = (size_t)MOE_S * MOE_D / 4;
        cast_bf16_kernel<<<(int)((n4 + 255) / 256), 256, 0, stream>>>(x, xb, n4);

        dim3 tb(32, 8);
        // gate: [D][H] -> rows [0,H) of [2H][D]
        transpose_cast_kernel<<<dim3(MOE_H / 32, MOE_D / 32, MOE_E), tb, 0, stream>>>(
            wg, wtGU, MOE_D, MOE_H,
            (size_t)MOE_D * MOE_H, (size_t)2 * MOE_H * MOE_D);
        // up: [D][H] -> rows [H,2H)
        transpose_cast_kernel<<<dim3(MOE_H / 32, MOE_D / 32, MOE_E), tb, 0, stream>>>(
            wu, wtGU + (size_t)MOE_H * MOE_D, MOE_D, MOE_H,
            (size_t)MOE_D * MOE_H, (size_t)2 * MOE_H * MOE_D);
        // down: [H][D] -> [D][H]
        transpose_cast_kernel<<<dim3(MOE_D / 32, MOE_H / 32, MOE_E), tb, 0, stream>>>(
            wd, wtDN, MOE_H, MOE_D,
            (size_t)MOE_H * MOE_D, (size_t)MOE_D * MOE_H);
    }

    // 3) experts: GEMM1 -> SwiGLU -> GEMM2 (+scaled combine into out)
    for (int e = 0; e < MOE_E; ++e) {
        gemm_bf16_kernel<0><<<dim3(2 * MOE_H / BN, MOE_S / BM), 128, 0, stream>>>(
            xb, wtGU + (size_t)e * 2 * MOE_H * MOE_D,
            MOE_S, 2 * MOE_H, MOE_D,
            Z, nullptr, nullptr, 0, 0);

        const size_t nact = (size_t)MOE_S * MOE_H;
        swiglu_kernel<<<(int)((nact + 255) / 256), 256, 0, stream>>>(Z, act);

        gemm_bf16_kernel<1><<<dim3(MOE_D / BN, MOE_S / BM), 128, 0, stream>>>(
            act, wtDN + (size_t)e * MOE_D * MOE_H,
            MOE_S, MOE_D, MOE_H,
            nullptr, out, cw, e, e > 0 ? 1 : 0);
    }
}
